// DGCNN_18494129177161
// MI455X (gfx1250) — compile-verified, matrix operand order checked by reading, not run on hardware
//
#include <hip/hip_runtime.h>
#include <hip/hip_bf16.h>

// ---------------- WMMA helpers (CDNA5 / gfx1250, wave32) ----------------
typedef __attribute__((ext_vector_type(16))) _Float16 v16h;
typedef __attribute__((ext_vector_type(8)))  _Float16 v8h;
typedef __attribute__((ext_vector_type(8)))  float    v8f;

// 16-bit A/B operand fragment load, per ISA 7.12.2 (16-bit A 16x32):
//   lane L: row = L&15, half = L>>4
//   VGPRs 0..3 hold K = 8*half .. 8*half+7   (one 16-byte chunk)
//   VGPRs 4..7 hold K = 16+8*half .. +7      (second 16-byte chunk)
__device__ __forceinline__ v16h load_frag(const _Float16* base, int row, int ld,
                                          int kbase, int half) {
  const _Float16* p = base + (size_t)row * ld + kbase + (half << 3);
  union { v16h v; v8h c[2]; } u;
  u.c[0] = *(const v8h*)(p);
  u.c[1] = *(const v8h*)(p + 16);
  return u.v;
}

__device__ __forceinline__ v8f wmma_f16(v16h a, v16h b, v8f c) {
  // (neg_a, A, neg_b, B, c_mod, C, reuse_a, reuse_b)
  return __builtin_amdgcn_wmma_f32_16x16x32_f16(false, a, false, b, (short)0, c,
                                                false, false);
}

__device__ __forceinline__ float lrelu(float z) { return z >= 0.f ? z : 0.01f * z; }

#define NEG_INF (-3.402823e38f)

// ---------------- weight f32 -> f16 (K-dim zero-padded) ----------------
__global__ void cvt_kernel(const float* __restrict__ src, _Float16* __restrict__ dst,
                           int rows, int Ks, int Kd) {
  int t = blockIdx.x * blockDim.x + threadIdx.x;
  int total = rows * Kd;
  if (t >= total) return;
  int r = t / Kd, j = t - r * Kd;
  dst[t] = (j < Ks) ? (_Float16)src[(size_t)r * Ks + j] : (_Float16)0.f;
}

// ---------------- kNN: one block (64 threads) per batch ----------------
// Hin: node-major features, row (b*64+c)*ld + j, f features per node.
__global__ void knn_kernel(const float* __restrict__ Hin, int ld, int f,
                           int* __restrict__ idx) {
  extern __shared__ __align__(16) char smem[];
  float* Xs   = (float*)smem;                               // 64*f
  float* rowb = (float*)(smem + (size_t)64 * f * 4);        // 64*64
  float* xx   = (float*)(smem + (size_t)64 * f * 4 + 64*64*4); // 64
  const int b = blockIdx.x;
  const int tid = threadIdx.x;  // 0..63
  const float* Hb = Hin + (size_t)b * 64 * ld;
  for (int e = tid; e < 64 * f; e += 64) {
    int c = e / f, j = e - c * f;
    Xs[c * f + j] = Hb[(size_t)c * ld + j];
  }
  __syncthreads();
  {
    const float* xi = Xs + tid * f;
    float s = 0.f;
    for (int j = 0; j < f; ++j) s += xi[j] * xi[j];
    xx[tid] = s;
  }
  __syncthreads();
  const float* xi = Xs + tid * f;
  const float xxi = xx[tid];
  for (int j2 = 0; j2 < 64; ++j2) {
    const float* xj = Xs + j2 * f;
    float s = 0.f;
    for (int j = 0; j < f; ++j) s += xi[j] * xj[j];
    rowb[tid * 64 + j2] = 2.f * s - xxi - xx[j2];  // neg squared distance
  }
  for (int t = 0; t < 20; ++t) {  // iterative argmax == top_k order
    float best = NEG_INF; int bj = 0;
    for (int j2 = 0; j2 < 64; ++j2) {
      float v = rowb[tid * 64 + j2];
      if (v > best) { best = v; bj = j2; }
    }
    idx[(b * 64 + tid) * 20 + t] = bj;
    rowb[tid * 64 + bj] = NEG_INF;
  }
}

// ---------------- EdgeConv: one block (128 threads) per (b, c) ----------------
// GEMM: M = d (out chans), N = 32 (k padded), K = Kpad (2f padded to %32).
__global__ void edgeconv_kernel(const float* __restrict__ Hin, int ld, int f,
                                const int* __restrict__ idx,
                                const _Float16* __restrict__ W16, int Kpad, int d,
                                const float* __restrict__ sc, const float* __restrict__ sh,
                                float* __restrict__ Hcat, _Float16* __restrict__ Hcat16,
                                int outOff) {
  extern __shared__ __align__(16) char smem[];
  _Float16* E  = (_Float16*)smem;                            // 32 * Kpad
  float* zbuf  = (float*)(smem + (size_t)32 * Kpad * 2);     // d * 32
  int* idxs    = (int*)(smem + (size_t)32 * Kpad * 2 + (size_t)d * 32 * 4);
  const int b = blockIdx.x >> 6;
  const int c = blockIdx.x & 63;
  const int tid = threadIdx.x;
  const float* hc = Hin + ((size_t)b * 64 + c) * ld;
  if (tid < 20) idxs[tid] = idx[(b * 64 + c) * 20 + tid];
  __syncthreads();
  // Stage padded edge features [k][2f] = cat(nb - ctr, ctr) as f16.
  const int total = 32 * Kpad;
  for (int e = tid; e < total; e += 128) {
    int k = e / Kpad, j = e - k * Kpad;
    _Float16 v = (_Float16)0.f;
    if (k < 20) {
      if (j < f) {
        const float* hn = Hin + ((size_t)b * 64 + idxs[k]) * ld;
        v = (_Float16)(hn[j] - hc[j]);
      } else if (j < 2 * f) {
        v = (_Float16)hc[j - f];
      }
    }
    E[e] = v;
  }
  __syncthreads();
  const int wave = tid >> 5, lane = tid & 31;
  const int half = lane >> 4, l16 = lane & 15;
  const int numTiles = (d >> 4) * 2;  // mtiles * 2 ntiles
  for (int t = wave; t < numTiles; t += 4) {
    const int mt = t >> 1, nt = t & 1;
    v8f acc = {0.f, 0.f, 0.f, 0.f, 0.f, 0.f, 0.f, 0.f};
    for (int kt = 0; kt < Kpad; kt += 32) {
      v16h a  = load_frag(W16, mt * 16 + l16, Kpad, kt, half);  // weights (global/L2)
      v16h bm = load_frag(E,   nt * 16 + l16, Kpad, kt, half);  // edge feats (LDS)
      acc = wmma_f16(a, bm, acc);
    }
    const int ncol = nt * 16 + l16;
    #pragma unroll
    for (int v = 0; v < 8; ++v) {
      int m = mt * 16 + v + 8 * half;  // D layout: VGPR v -> row v + 8*half
      zbuf[m * 32 + ncol] = lrelu(acc[v] * sc[m] + sh[m]);
    }
  }
  __syncthreads();
  for (int m = tid; m < d; m += 128) {  // max over real neighbors only
    float vmax = zbuf[m * 32];
    for (int k = 1; k < 20; ++k) vmax = fmaxf(vmax, zbuf[m * 32 + k]);
    size_t o = ((size_t)b * 64 + c) * 512 + outOff + m;
    Hcat[o]   = vmax;
    Hcat16[o] = (_Float16)vmax;
  }
}

// ---------------- global point-wise conv + max/mean pooling ----------------
// grid = B * 16; each block: 64 output rows x 64 nodes, K = 512.
__global__ void gp_kernel(const _Float16* __restrict__ Hcat16,
                          const _Float16* __restrict__ GPW,
                          const float* __restrict__ gsc, const float* __restrict__ gsh,
                          _Float16* __restrict__ feat16) {
  __shared__ float zbuf[64 * 64];
  const int b  = blockIdx.x >> 4;
  const int m0 = (blockIdx.x & 15) * 64;
  const int tid = threadIdx.x;
  const int wave = tid >> 5, lane = tid & 31;
  const int half = lane >> 4, l16 = lane & 15;
  const int mt = m0 + wave * 16;
  const _Float16* Bb = Hcat16 + (size_t)b * 64 * 512;
  for (int nt = 0; nt < 4; ++nt) {
    v8f acc = {0.f, 0.f, 0.f, 0.f, 0.f, 0.f, 0.f, 0.f};
    for (int kt = 0; kt < 512; kt += 32) {
      v16h a  = load_frag(GPW, mt + l16, 512, kt, half);
      v16h bm = load_frag(Bb,  nt * 16 + l16, 512, kt, half);
      acc = wmma_f16(a, bm, acc);
    }
    const int ncol = nt * 16 + l16;
    #pragma unroll
    for (int v = 0; v < 8; ++v) {
      int m = mt + v + 8 * half;
      zbuf[(m - m0) * 64 + ncol] = lrelu(acc[v] * gsc[m] + gsh[m]);
    }
  }
  __syncthreads();
  if (tid < 64) {
    const int m = m0 + tid;
    float vmax = NEG_INF, vsum = 0.f;
    for (int c = 0; c < 64; ++c) {
      float z = zbuf[tid * 64 + c];
      vmax = fmaxf(vmax, z);
      vsum += z;
    }
    feat16[b * 2048 + m]        = (_Float16)vmax;
    feat16[b * 2048 + 1024 + m] = (_Float16)(vsum * (1.f / 64.f));
  }
}

// ---------------- MLP layer: out[b][m] = lrelu((in @ W^T)*sc + bi) ----------------
// N = 64 batches, M = blockDim-of-64 rows per block, K = Kd.
__global__ void mlp_kernel(const _Float16* __restrict__ In16, int Kd,
                           const _Float16* __restrict__ W16, int M,
                           const float* __restrict__ sc, const float* __restrict__ bi,
                           _Float16* __restrict__ out16, float* __restrict__ outf) {
  const int m0 = blockIdx.x * 64;
  const int tid = threadIdx.x;
  const int wave = tid >> 5, lane = tid & 31;
  const int half = lane >> 4, l16 = lane & 15;
  const int mt = m0 + wave * 16;
  for (int nt = 0; nt < 4; ++nt) {
    v8f acc = {0.f, 0.f, 0.f, 0.f, 0.f, 0.f, 0.f, 0.f};
    for (int kt = 0; kt < Kd; kt += 32) {
      v16h a  = load_frag(W16,  mt + l16, Kd, kt, half);
      v16h bm = load_frag(In16, nt * 16 + l16, Kd, kt, half);
      acc = wmma_f16(a, bm, acc);
    }
    const int n = nt * 16 + l16;  // batch index
    #pragma unroll
    for (int v = 0; v < 8; ++v) {
      int m = mt + v + 8 * half;
      float z = lrelu(acc[v] * sc[m] + bi[m]);
      if (out16) out16[n * M + m] = (_Float16)z;
      if (outf)  outf[n * M + m]  = z;
    }
  }
}

// ---------------- tiny 256 -> 4 head ----------------
__global__ void head_kernel(const float* __restrict__ h2, const float* __restrict__ w3,
                            const float* __restrict__ b3, float* __restrict__ out) {
  int t = blockIdx.x * blockDim.x + threadIdx.x;
  if (t >= 256) return;
  int b = t >> 2, o = t & 3;
  float acc = b3[o];
  const float* hb = h2 + b * 256;
  const float* wr = w3 + o * 256;
  for (int j = 0; j < 256; ++j) acc += hb[j] * wr[j];
  out[t] = acc;
}

// ---------------- host orchestration ----------------
extern "C" void kernel_launch(void* const* d_in, const int* in_sizes, int n_in,
                              void* d_out, int out_size, void* d_ws, size_t ws_size,
                              hipStream_t stream) {
  (void)in_sizes; (void)n_in; (void)out_size; (void)ws_size;
  const float* x     = (const float*)d_in[0];        // (64,64,500)
  const float* ecw[4]   = {(const float*)d_in[1], (const float*)d_in[2],
                           (const float*)d_in[3], (const float*)d_in[4]};
  const float* ecsc[4]  = {(const float*)d_in[5], (const float*)d_in[6],
                           (const float*)d_in[7], (const float*)d_in[8]};
  const float* ecsh[4]  = {(const float*)d_in[9], (const float*)d_in[10],
                           (const float*)d_in[11], (const float*)d_in[12]};
  const float* gp_w  = (const float*)d_in[13];       // (1024,512)
  const float* gp_sc = (const float*)d_in[14];
  const float* gp_sh = (const float*)d_in[15];
  const float* w1 = (const float*)d_in[16];          // (512,2048)
  const float* s1 = (const float*)d_in[17];
  const float* b1 = (const float*)d_in[18];
  const float* w2 = (const float*)d_in[19];          // (256,512)
  const float* s2 = (const float*)d_in[20];
  const float* b2 = (const float*)d_in[21];
  const float* w3 = (const float*)d_in[22];          // (4,256)
  const float* b3 = (const float*)d_in[23];

  char* ws = (char*)d_ws;
  size_t off = 0;
  auto carve = [&](size_t bytes) -> char* {
    char* p = ws + off;
    off = (off + bytes + 255) & ~(size_t)255;
    return p;
  };
  _Float16* W16_0  = (_Float16*)carve((size_t)64 * 1024 * 2);
  _Float16* W16_1  = (_Float16*)carve((size_t)64 * 128 * 2);
  _Float16* W16_2  = (_Float16*)carve((size_t)128 * 128 * 2);
  _Float16* W16_3  = (_Float16*)carve((size_t)256 * 256 * 2);
  _Float16* GPW16  = (_Float16*)carve((size_t)1024 * 512 * 2);
  _Float16* W1_16  = (_Float16*)carve((size_t)512 * 2048 * 2);
  _Float16* W2_16  = (_Float16*)carve((size_t)256 * 512 * 2);
  float*    Hcat   = (float*)   carve((size_t)64 * 64 * 512 * 4);
  _Float16* Hcat16 = (_Float16*)carve((size_t)64 * 64 * 512 * 2);
  int*      idx    = (int*)     carve((size_t)64 * 64 * 20 * 4);
  _Float16* feat16 = (_Float16*)carve((size_t)64 * 2048 * 2);
  _Float16* h1_16  = (_Float16*)carve((size_t)64 * 512 * 2);
  float*    h2_f   = (float*)   carve((size_t)64 * 256 * 4);

  auto cvt = [&](const float* s, _Float16* d, int rows, int Ks, int Kd) {
    int total = rows * Kd;
    cvt_kernel<<<(total + 255) / 256, 256, 0, stream>>>(s, d, rows, Ks, Kd);
  };
  cvt(ecw[0], W16_0, 64, 1000, 1024);
  cvt(ecw[1], W16_1, 64, 128, 128);
  cvt(ecw[2], W16_2, 128, 128, 128);
  cvt(ecw[3], W16_3, 256, 256, 256);
  cvt(gp_w, GPW16, 1024, 512, 512);
  cvt(w1, W1_16, 512, 2048, 2048);
  cvt(w2, W2_16, 256, 512, 512);

  // EdgeConv blocks: {input base, ld, f, W16, Kpad, d, outOff}
  struct Layer { const float* in; int ld, f; const _Float16* w; int Kpad, d, outOff; };
  Layer L[4] = {
    { x,          500, 500, W16_0, 1024,  64,   0 },
    { Hcat + 0,   512,  64, W16_1,  128,  64,  64 },
    { Hcat + 64,  512,  64, W16_2,  128, 128, 128 },
    { Hcat + 128, 512, 128, W16_3,  256, 256, 256 },
  };
  for (int i = 0; i < 4; ++i) {
    size_t knnS = (size_t)64 * L[i].f * 4 + 64 * 64 * 4 + 64 * 4;
    knn_kernel<<<64, 64, knnS, stream>>>(L[i].in, L[i].ld, L[i].f, idx);
    size_t ecS = (size_t)32 * L[i].Kpad * 2 + (size_t)L[i].d * 32 * 4 + 32 * 4;
    edgeconv_kernel<<<64 * 64, 128, ecS, stream>>>(
        L[i].in, L[i].ld, L[i].f, idx, L[i].w, L[i].Kpad, L[i].d,
        ecsc[i], ecsh[i], Hcat, Hcat16, L[i].outOff);
  }

  gp_kernel<<<64 * 16, 128, 0, stream>>>(Hcat16, GPW16, gp_sc, gp_sh, feat16);

  mlp_kernel<<<512 / 64, 128, 0, stream>>>(feat16, 2048, W1_16, 512, s1, b1,
                                           h1_16, (float*)nullptr);
  mlp_kernel<<<256 / 64, 128, 0, stream>>>(h1_16, 512, W2_16, 256, s2, b2,
                                           (_Float16*)nullptr, h2_f);

  head_kernel<<<1, 256, 0, stream>>>(h2_f, w3, b3, (float*)d_out);
}
